// SelfAttention_2619930051042
// MI455X (gfx1250) — matrix operand order repaired and checked by reading, not compiled, not run on hardware
//
#include <hip/hip_runtime.h>
#include <hip/hip_bf16.h>

// ---------------------------------------------------------------------------
// XCA channel-attention for MI455X (gfx1250), bf16 WMMA pipeline.
//   0) zero norm/G accumulators and vT pad columns
//   1) qkv = Wqkv @ x (bf16 WMMA, double-buffered LDS X-tile); q/k stored
//      [ch][n] via LDS-transposed coalesced b128 stores + row sum-of-squares;
//      v stored transposed vT[b][n][head*64+d] with packed b128 stores.
//   2) G = q @ k^T (bf16 WMMA, LDS-free, K split across blocks, f32 atomics)
//   2b) attn = softmax(G / (|q||k|) * temp), bf16, K padded to 64
//   3) y = Wout @ (attn @ v) fused, LDS-free: v B-frags direct from vT,
//      t kept in registers, C-layout -> B-layout via shfl_xor(16)
// ---------------------------------------------------------------------------

typedef __attribute__((ext_vector_type(16))) __bf16 v16bf;
typedef __attribute__((ext_vector_type(8)))  float  v8f;

struct alignas(16) B8 { __bf16 h[8]; };

#define NPIX 16384
#define CIN  192
#define XSTRIDE 40        // halves per LDS X-tile row (80 B, 16B multiple)
#define XBUF (128 * XSTRIDE)

__device__ inline v16bf mk16(const B8& lo, const B8& hi) {
  v16bf r;
#pragma unroll
  for (int i = 0; i < 8; i++) { r[i] = lo.h[i]; r[i + 8] = hi.h[i]; }
  return r;
}

__device__ inline v16bf mk16f(const float4& a, const float4& b,
                              const float4& c, const float4& d) {
  v16bf r;
  r[0]=(__bf16)a.x; r[1]=(__bf16)a.y; r[2]=(__bf16)a.z; r[3]=(__bf16)a.w;
  r[4]=(__bf16)b.x; r[5]=(__bf16)b.y; r[6]=(__bf16)b.z; r[7]=(__bf16)b.w;
  r[8]=(__bf16)c.x; r[9]=(__bf16)c.y; r[10]=(__bf16)c.z; r[11]=(__bf16)c.w;
  r[12]=(__bf16)d.x; r[13]=(__bf16)d.y; r[14]=(__bf16)d.z; r[15]=(__bf16)d.w;
  return r;
}

__device__ inline v8f wmma_bf16(v16bf a, v16bf b, v8f c) {
  return __builtin_amdgcn_wmma_f32_16x16x32_bf16(false, a, false, b,
                                                 (short)0, c, false, false);
}

__device__ inline B8 pack8(const v8f& a) {
  B8 r;
#pragma unroll
  for (int i = 0; i < 8; i++) r.h[i] = (__bf16)a[i];
  return r;
}

union U8 { B8 b; unsigned int u[4]; };
__device__ inline B8 shfl16(const B8& v) {
  U8 x; x.b = v;
#pragma unroll
  for (int i = 0; i < 4; i++) x.u[i] = __shfl_xor(x.u[i], 16, 32);
  return x.b;
}

// ---------------------------------------------------------------------------
// Stage 0: init
// ---------------------------------------------------------------------------
__global__ void zero_f32_k(float* p, int n) {
  int i = blockIdx.x * blockDim.x + threadIdx.x;
  if (i < n) p[i] = 0.f;
}

// zero vT pad columns: vT[b][n][hd*64 + 48 .. 63]
__global__ void zero_vpad_k(__bf16* vT) {
  int i = blockIdx.x * blockDim.x + threadIdx.x;   // 0 .. 524287
  int hd = i & 3;
  int n  = (i >> 2) & 16383;
  int b  = i >> 16;
  size_t off = ((size_t)(b * NPIX + n)) * 256 + hd * 64 + 48;
  B8 z{};
  *reinterpret_cast<B8*>(vT + off)     = z;
  *reinterpret_cast<B8*>(vT + off + 8) = z;
}

// ---------------------------------------------------------------------------
// Stage 1: qkv GEMM.  C[576,16384] = Wqkv[576,192] x X[192,16384] per batch.
// Block 256 = 8 waves; tile 64M x 128N; wave -> (wm, wn) = 16M x 64N (4 subs).
// Double-buffered LDS X tile; one barrier per k-step.
// ---------------------------------------------------------------------------
__global__ __launch_bounds__(256) void qkv_k(
    const float* __restrict__ x, const float* __restrict__ Wqkv,
    __bf16* __restrict__ qk, __bf16* __restrict__ vT,
    float* __restrict__ nrm2) {
  const int b = blockIdx.z;
  const int m0 = blockIdx.y * 64;
  const int n0 = blockIdx.x * 128;
  const int tid = threadIdx.x, lane = tid & 31, wave = tid >> 5;
  const int wm = wave >> 1, wn = wave & 1;

  // xs: 2 x X-tile [n][k] bf16; ts: C tile [m][n] stride 136 for writeout.
  __shared__ union { __bf16 xs[2 * XBUF]; __bf16 ts[64 * 136]; } sh;
  const float* xb = x + (size_t)b * CIN * NPIX;

  const int lkk = tid >> 5;          // loader: k row base within pass
  const int lj  = (tid & 31) << 2;   // loader: n column (4 floats)

  v8f acc[4] = {};

  // prologue: fill buffer 0 with k-step 0
  {
    float4 f[4];
#pragma unroll
    for (int p = 0; p < 4; p++)
      f[p] = *reinterpret_cast<const float4*>(
          xb + (size_t)(p * 8 + lkk) * NPIX + n0 + lj);
#pragma unroll
    for (int p = 0; p < 4; p++) {
      int k = p * 8 + lkk;
      sh.xs[(lj + 0) * XSTRIDE + k] = (__bf16)f[p].x;
      sh.xs[(lj + 1) * XSTRIDE + k] = (__bf16)f[p].y;
      sh.xs[(lj + 2) * XSTRIDE + k] = (__bf16)f[p].z;
      sh.xs[(lj + 3) * XSTRIDE + k] = (__bf16)f[p].w;
    }
  }
  __syncthreads();

#pragma unroll
  for (int k0 = 0; k0 < CIN; k0 += 32) {
    const int buf = (k0 >> 5) & 1;
    const bool more = (k0 + 32) < CIN;

    // issue next tile's global loads early (latency hidden under WMMAs)
    float4 fn[4];
    if (more) {
#pragma unroll
      for (int p = 0; p < 4; p++)
        fn[p] = *reinterpret_cast<const float4*>(
            xb + (size_t)(k0 + 32 + p * 8 + lkk) * NPIX + n0 + lj);
    }

    // A fragment: row m, halves [kb..kb+7] and [kb+16..kb+23]
    int arow = m0 + wm * 16 + (lane & 15);
    const float4* wq = reinterpret_cast<const float4*>(
        Wqkv + (size_t)arow * CIN + k0 + ((lane >> 4) << 3));
    v16bf av = mk16f(wq[0], wq[1], wq[4], wq[5]);

    // batch all four B fragments, then the four WMMAs
    const __bf16* xs = sh.xs + buf * XBUF;
    v16bf bv[4];
#pragma unroll
    for (int s = 0; s < 4; s++) {
      int ncol = wn * 64 + s * 16 + (lane & 15);
      const __bf16* bp = &xs[ncol * XSTRIDE + ((lane >> 4) << 3)];
      bv[s] = mk16(*reinterpret_cast<const B8*>(bp),
                   *reinterpret_cast<const B8*>(bp + 16));
    }
#pragma unroll
    for (int s = 0; s < 4; s++)
      acc[s] = wmma_bf16(av, bv[s], acc[s]);

    // convert + store next tile into the other buffer
    if (more) {
      __bf16* xw = sh.xs + (buf ^ 1) * XBUF;
#pragma unroll
      for (int p = 0; p < 4; p++) {
        int k = p * 8 + lkk;
        xw[(lj + 0) * XSTRIDE + k] = (__bf16)fn[p].x;
        xw[(lj + 1) * XSTRIDE + k] = (__bf16)fn[p].y;
        xw[(lj + 2) * XSTRIDE + k] = (__bf16)fn[p].z;
        xw[(lj + 3) * XSTRIDE + k] = (__bf16)fn[p].w;
      }
    }
    __syncthreads();
  }

  if (m0 < 384) {
    // -------- q/k path: sum-of-squares + LDS transpose + coalesced stores
    const int mloc = wm * 16 + ((lane & 16) ? 8 : 0);
    float s2[8] = {0.f,0.f,0.f,0.f,0.f,0.f,0.f,0.f};
#pragma unroll
    for (int s = 0; s < 4; s++) {
      int ncol = wn * 64 + s * 16 + (lane & 15);
#pragma unroll
      for (int r = 0; r < 8; r++) {
        float vv = acc[s][r];
        s2[r] += vv * vv;
        sh.ts[(mloc + r) * 136 + ncol] = (__bf16)vv;
      }
    }
#pragma unroll
    for (int r = 0; r < 8; r++) {
      float v = s2[r];
      v += __shfl_xor(v, 1, 32);
      v += __shfl_xor(v, 2, 32);
      v += __shfl_xor(v, 4, 32);
      v += __shfl_xor(v, 8, 32);
      if ((lane & 15) == 0)
        atomicAdd(&nrm2[b * 384 + m0 + mloc + r], v);
    }
    __syncthreads();
    // coalesced b128 writeout of the 64x128 tile
#pragma unroll
    for (int pp = 0; pp < 4; pp++) {
      int row = ((pp & 1) << 5) + (tid >> 3);
      int j   = ((tid & 7) << 3) + ((pp >> 1) << 6);
      B8 vv = *reinterpret_cast<const B8*>(&sh.ts[row * 136 + j]);
      *reinterpret_cast<B8*>(qk + ((size_t)b * 384 + m0 + row) * NPIX + n0 + j) = vv;
    }
  } else {
    // -------- v path: packed transposed stores into vT[b][n][hd*64+d]
    const int m0v   = m0 + wm * 16 - 384;          // uniform per wave
    const int hd    = m0v / 48;
    const int dbase = (m0v % 48) + ((lane & 16) ? 8 : 0);
#pragma unroll
    for (int s = 0; s < 4; s++) {
      int n = n0 + wn * 64 + s * 16 + (lane & 15);
      B8 pv = pack8(acc[s]);
      *reinterpret_cast<B8*>(vT + ((size_t)(b * NPIX + n)) * 256 + hd * 64 + dbase) = pv;
    }
  }
}

// ---------------------------------------------------------------------------
// Stage 2: G[b,h,48,48] += q[48,Kslice] @ k[48,Kslice]^T.  LDS-free.
// ---------------------------------------------------------------------------
__global__ __launch_bounds__(288) void gram_k(const __bf16* __restrict__ qk,
                                              float* __restrict__ G) {
  const int b = blockIdx.z, hd = blockIdx.y;
  const int kbase = blockIdx.x * 512;
  const int wave = threadIdx.x >> 5, lane = threadIdx.x & 31;
  const int mi = wave / 3, ni = wave % 3;

  const __bf16* qrow = qk + ((size_t)b * 384 + hd * 48 + mi * 16) * NPIX;
  const __bf16* krow = qk + ((size_t)b * 384 + 192 + hd * 48 + ni * 16) * NPIX;

  v8f acc = {};
#pragma unroll 4
  for (int ks = 0; ks < 16; ks++) {
    int k0 = kbase + ks * 32;
    const __bf16* ap = qrow + (size_t)(lane & 15) * NPIX + k0 + ((lane >> 4) << 3);
    __builtin_prefetch(ap + 256, 0, 0);          // ~512B ahead in the stream
    v16bf av = mk16(*reinterpret_cast<const B8*>(ap),
                    *reinterpret_cast<const B8*>(ap + 16));
    const __bf16* bp = krow + (size_t)(lane & 15) * NPIX + k0 + ((lane >> 4) << 3);
    __builtin_prefetch(bp + 256, 0, 0);
    v16bf bv = mk16(*reinterpret_cast<const B8*>(bp),
                    *reinterpret_cast<const B8*>(bp + 16));
    acc = wmma_bf16(av, bv, acc);
  }

  float* gp = G + ((size_t)b * 4 + hd) * 48 * 48;
  const int mrow = mi * 16 + ((lane & 16) ? 8 : 0);
  const int ncol = ni * 16 + (lane & 15);
#pragma unroll
  for (int r = 0; r < 8; r++)
    atomicAdd(&gp[(mrow + r) * 48 + ncol], acc[r]);
}

// ---------------------------------------------------------------------------
// Stage 2b: fold norms + temperature, softmax over 48, bf16 padded to K=64.
// ---------------------------------------------------------------------------
__global__ __launch_bounds__(32) void softmax_k(
    const float* __restrict__ G, const float* __restrict__ nrm2,
    const float* __restrict__ temp, __bf16* __restrict__ attnb) {
  const int row = blockIdx.x;             // [0, b*4*48)
  const int c = row % 48; const int bh = row / 48;
  const int hd = bh & 3; const int b = bh >> 2;
  const int lane = threadIdx.x;
  const float* g = G + (size_t)row * 48;
  const float* nq = nrm2 + b * 384;
  const float qn = fmaxf(sqrtf(nq[hd * 48 + c]), 1e-12f);
  const float tv = temp[hd];

  int d0 = lane, d1 = lane + 32;
  float kn0 = fmaxf(sqrtf(nq[192 + hd * 48 + d0]), 1e-12f);
  float s0 = g[d0] / (qn * kn0) * tv;
  float s1 = -3.0e38f;
  if (d1 < 48) {
    float kn1 = fmaxf(sqrtf(nq[192 + hd * 48 + d1]), 1e-12f);
    s1 = g[d1] / (qn * kn1) * tv;
  }
  float m = fmaxf(s0, s1);
#pragma unroll
  for (int msk = 16; msk >= 1; msk >>= 1) m = fmaxf(m, __shfl_xor(m, msk, 32));
  float e0 = expf(s0 - m);
  float e1 = (d1 < 48) ? expf(s1 - m) : 0.f;
  float sum = e0 + e1;
#pragma unroll
  for (int msk = 16; msk >= 1; msk >>= 1) sum += __shfl_xor(sum, msk, 32);
  float inv = 1.f / sum;

  __bf16* ap = attnb + (size_t)row * 64;
  ap[d0] = (__bf16)(e0 * inv);
  ap[d1] = (d1 < 48) ? (__bf16)(e1 * inv) : (__bf16)0.f;
}

// ---------------------------------------------------------------------------
// Stage 3: y = Wout @ (attn @ v), fused, LDS-free.  Block 256 = 8 waves,
// n-tile 128; wave owns one 16-wide n-subtile.  v B-frags direct from vT
// (hoisted across mi); t = attn@v lives in registers as 12 packed bf16x8;
// C-layout -> B-fragment for GEMM2 via shfl_xor(16).
// ---------------------------------------------------------------------------
__global__ __launch_bounds__(256) void attnout_k(
    const __bf16* __restrict__ vT, const __bf16* __restrict__ attnb,
    const float* __restrict__ Wout, float* __restrict__ out) {
  const int b = blockIdx.y;
  const int n0 = blockIdx.x * 128;
  const int tid = threadIdx.x, lane = tid & 31, w = tid >> 5;

  B8 tpk[12];
  const __bf16* vbase =
      vT + ((size_t)(b * NPIX + n0 + w * 16 + (lane & 15))) * 256;

#pragma unroll
  for (int hd = 0; hd < 4; hd++) {
    // B fragments for this head (shared across the 3 mi tiles)
    const __bf16* bp0 = vbase + hd * 64 + ((lane >> 4) << 3);
    v16bf bv0 = mk16(*reinterpret_cast<const B8*>(bp0),
                     *reinterpret_cast<const B8*>(bp0 + 16));
    const __bf16* bp1 = bp0 + 32;
    v16bf bv1 = mk16(*reinterpret_cast<const B8*>(bp1),
                     *reinterpret_cast<const B8*>(bp1 + 16));
#pragma unroll
    for (int mi = 0; mi < 3; mi++) {
      v8f acc = {};
      const __bf16* ap = attnb +
          ((size_t)((b * 4 + hd) * 48) + mi * 16 + (lane & 15)) * 64 +
          ((lane >> 4) << 3);
      v16bf av0 = mk16(*reinterpret_cast<const B8*>(ap),
                       *reinterpret_cast<const B8*>(ap + 16));
      v16bf av1 = mk16(*reinterpret_cast<const B8*>(ap + 32),
                       *reinterpret_cast<const B8*>(ap + 48));
      acc = wmma_bf16(av0, bv0, acc);
      acc = wmma_bf16(av1, bv1, acc);
      tpk[hd * 3 + mi] = pack8(acc);
    }
  }

  // GEMM2: y[192,128] = Wout[192,192] @ t[192,128]; t from registers.
  const int nglob = n0 + w * 16 + (lane & 15);
  for (int mo = 0; mo < 12; mo++) {
    v8f y = {};
#pragma unroll
    for (int ks = 0; ks < 6; ks++) {
      int arow = mo * 16 + (lane & 15);
      const float4* wq = reinterpret_cast<const float4*>(
          Wout + (size_t)arow * CIN + ks * 32 + ((lane >> 4) << 4));
      v16bf av = mk16f(wq[0], wq[1], wq[2], wq[3]);
      B8 o0 = shfl16(tpk[2 * ks]);
      B8 o1 = shfl16(tpk[2 * ks + 1]);
      B8 blo = (lane < 16) ? tpk[2 * ks] : o1;
      B8 bhi = (lane < 16) ? o0 : tpk[2 * ks + 1];
      v16bf bv = mk16(blo, bhi);
      y = wmma_bf16(av, bv, y);
    }
    int obase = mo * 16 + ((lane & 16) ? 8 : 0);
#pragma unroll
    for (int r = 0; r < 8; r++)
      out[((size_t)b * CIN + obase + r) * NPIX + nglob] = y[r];
  }
}

// ---------------------------------------------------------------------------
// Workspace layout (bytes):
//   qk    [8][384][16384] bf16 : 100,663,296
//   vT    [8][16384][256] bf16 (cols hd*64+48..63 zero) : 67,108,864
//   nrm2  [8][384] f32 : 12,288
//   G     [8][4][48][48] f32 : 294,912
//   attnb [8][4][48][64] bf16 : 196,608     (total ~160.5 MiB)
// ---------------------------------------------------------------------------
extern "C" void kernel_launch(void* const* d_in, const int* in_sizes, int n_in,
                              void* d_out, int out_size, void* d_ws, size_t ws_size,
                              hipStream_t stream) {
  const float* x    = (const float*)d_in[0];
  const float* Wqkv = (const float*)d_in[1];
  const float* Wout = (const float*)d_in[2];
  const float* temp = (const float*)d_in[3];
  float* out = (float*)d_out;

  char* ws = (char*)d_ws;
  __bf16* qk    = (__bf16*)(ws);
  __bf16* vT    = (__bf16*)(ws + 100663296);
  float*  nrm2  = (float*) (ws + 167772160);
  float*  G     = (float*) (ws + 167784448);
  __bf16* attnb = (__bf16*)(ws + 168079360);

  // Stage 0: zero accumulators (nrm2 + G contiguous) and vT pad columns.
  zero_f32_k<<<dim3((76800 + 255) / 256), 256, 0, stream>>>(nrm2, 76800);
  zero_vpad_k<<<dim3(2048), 256, 0, stream>>>(vT);

  // Stage 1: qkv projection + row sum-of-squares.
  qkv_k<<<dim3(128, 9, 8), 256, 0, stream>>>(x, Wqkv, qk, vT, nrm2);

  // Stage 2: Gram matrix, K split 32 ways.
  gram_k<<<dim3(32, 4, 8), 288, 0, stream>>>(qk, G);

  // Stage 2b: normalize + softmax -> bf16 attn (K padded to 64).
  softmax_k<<<dim3(8 * 4 * 48), 32, 0, stream>>>(G, nrm2, temp, attnb);

  // Stage 3: fused attn@v and output projection.
  attnout_k<<<dim3(128, 8), 256, 0, stream>>>(vT, attnb, Wout, out);
}
